// GATModel_org_75651553951877
// MI455X (gfx1250) — matrix operand order repaired and checked by reading, not compiled, not run on hardware
//
#include <hip/hip_runtime.h>
#include <hip/hip_bf16.h>

typedef __bf16 bf16;
typedef __attribute__((ext_vector_type(16))) __bf16 v16bf;
typedef __attribute__((ext_vector_type(2)))  __bf16 v2bf;
typedef __attribute__((ext_vector_type(8)))  float  v8f;
typedef __attribute__((ext_vector_type(4)))  float  v4f;
typedef __attribute__((ext_vector_type(4)))  int    v4i;

#define NN    8192
#define NF    512
#define NH    128
#define NCL   40
#define NWORD 256   // NN/32 mask words per row

__device__ __forceinline__ unsigned pk2bf(float a, float b) {
#if __has_builtin(__builtin_amdgcn_cvt_pk_bf16_f32)
  union { v2bf h; unsigned u; } x;
  x.h = __builtin_amdgcn_cvt_pk_bf16_f32(a, b);   // single v_cvt_pk_bf16_f32
  return x.u;
#else
  union { bf16 h[2]; unsigned u; } x;
  x.h[0] = (bf16)a; x.h[1] = (bf16)b;
  return x.u;
#endif
}

union FragBF { unsigned u[8]; v4i i4[2]; v16bf v; };

// ---------------------------------------------------------------------------
// 1. Pack adjacency [8192 x 8192] int32 -> bitmask [8192 x 256] u32 (8 MB, L2-resident)
// ---------------------------------------------------------------------------
__global__ __launch_bounds__(256) void pack_adj(const int* __restrict__ adj,
                                                unsigned* __restrict__ mask) {
  int lane = threadIdx.x & 31;
  int word = blockIdx.x * 8 + (threadIdx.x >> 5);       // 8 waves -> 8 words/block
  int a = adj[(size_t)word * 32 + lane];                // coalesced 128B per wave
  unsigned long long b = __ballot(a > 0);               // wave32: low 32 bits
  if (lane == 0) mask[word] = (unsigned)b;
}

// ---------------------------------------------------------------------------
// 2. GEMM: C[8192 x ldc] = concat(A1[K1], A2[K2]) @ W[(K1+K2) x Nout] (+bias)
//    bf16 WMMA, f32 accumulate, K-chunks of 64.
//    Block: 256 threads = 8 waves; wave w owns rows 16w..16w+15 and all NT col tiles.
// ---------------------------------------------------------------------------
template<int NT>
__global__ __launch_bounds__(256) void gemm_bf16(
    const float* __restrict__ A1, int K1,
    const float* __restrict__ A2, int K2,
    const float* __restrict__ W,  int ldw,
    const float* __restrict__ bias,
    float* __restrict__ Cf, bf16* __restrict__ Cb, int Nout) {
  const int ldc = NT * 16;
  __shared__ bf16 Ws[NT * 16][72];   // column-major W chunk: [n][k], k=0..63, padded

  const int tid  = threadIdx.x;
  const int L    = tid & 31;
  const int w    = tid >> 5;
  const int half = L >> 4;
  const int r    = blockIdx.x * 128 + w * 16 + (L & 15);   // A row for this lane
  const int kchunks = (K1 + K2) >> 6;

  v8f acc[NT];
  #pragma unroll
  for (int c = 0; c < NT; ++c)
    #pragma unroll
    for (int e = 0; e < 8; ++e) acc[c][e] = 0.f;

  const bf16* brow = &Ws[L & 15][8 * half];    // per-lane base inside Ws

  for (int kc = 0; kc < kchunks; ++kc) {
    const int kbase = kc << 6;
    __syncthreads();
    // stage W chunk: pack 2 consecutive-k bf16 per u32, single ds_store_b32
    for (int idx = tid; idx < ldc * 32; idx += 256) {
      int n = idx % ldc, k = (idx / ldc) * 2;
      float v0 = 0.f, v1 = 0.f;
      if (n < Nout) {
        v0 = W[(size_t)(kbase + k) * ldw + n];
        v1 = W[(size_t)(kbase + k + 1) * ldw + n];
      }
      *(unsigned*)&Ws[n][k] = pk2bf(v0, v1);
    }
    __syncthreads();

    // A fragments (two 16x32 k-halves): 32 f32 from global -> 16 packed cvt
    const float* Asrc; int kloc;
    if (kbase < K1) { Asrc = A1 + (size_t)r * K1; kloc = kbase; }
    else            { Asrc = A2 + (size_t)r * K2; kloc = kbase - K1; }
    FragBF av[2];
    #pragma unroll
    for (int kf = 0; kf < 2; ++kf) {
      const float* p = Asrc + kloc + 32 * kf;
      v4f a0 = *(const v4f*)(p + 8 * half);
      v4f a1 = *(const v4f*)(p + 8 * half + 4);
      v4f a2 = *(const v4f*)(p + 16 + 8 * half);
      v4f a3 = *(const v4f*)(p + 16 + 8 * half + 4);
      av[kf].u[0] = pk2bf(a0[0], a0[1]);
      av[kf].u[1] = pk2bf(a0[2], a0[3]);
      av[kf].u[2] = pk2bf(a1[0], a1[1]);
      av[kf].u[3] = pk2bf(a1[2], a1[3]);
      av[kf].u[4] = pk2bf(a2[0], a2[1]);
      av[kf].u[5] = pk2bf(a2[2], a2[3]);
      av[kf].u[6] = pk2bf(a3[0], a3[1]);
      av[kf].u[7] = pk2bf(a3[2], a3[3]);
    }

    // (c, kf) pipeline; scheduler overlaps the two ds_load_b128 per fragment
    FragBF ub[2];
    ub[0].i4[0] = *(const v4i*)(brow);
    ub[0].i4[1] = *(const v4i*)(brow + 16);
    #pragma unroll
    for (int t = 0; t < 2 * NT; ++t) {
      int nt = t + 1;
      if (nt < 2 * NT) {
        const bf16* bp = brow + (size_t)(nt >> 1) * 16 * 72 + 32 * (nt & 1);
        ub[nt & 1].i4[0] = *(const v4i*)bp;
        ub[nt & 1].i4[1] = *(const v4i*)(bp + 16);
      }
      acc[t >> 1] = __builtin_amdgcn_wmma_f32_16x16x32_bf16(
          false, av[t & 1].v, false, ub[t & 1].v, (short)0, acc[t >> 1], false, false);
    }
  }

  // store: lane L -> col n = c*16 + (L&15), rows M = 8*half + v
  #pragma unroll
  for (int c = 0; c < NT; ++c) {
    int n = c * 16 + (L & 15);
    float bv = (bias && n < Nout) ? bias[n] : 0.f;
    #pragma unroll
    for (int v = 0; v < 8; ++v) {
      size_t ro = (size_t)(blockIdx.x * 128 + w * 16 + 8 * half + v) * ldc + n;
      float val = acc[c][v] + bv;
      if (Cf) Cf[ro] = val;
      if (Cb) Cb[ro] = (__bf16)val;
    }
  }
}

// ---------------------------------------------------------------------------
// 3. Row dots: s[i] = Wh[i,:].a_src, d[i] = Wh[i,:].a_dst   (wave per row)
// ---------------------------------------------------------------------------
__global__ __launch_bounds__(256) void rowdots(const float* __restrict__ Wh, int ldc, int Nout,
                                               const float* __restrict__ asrc,
                                               const float* __restrict__ adst,
                                               float* __restrict__ s, float* __restrict__ d) {
  int L = threadIdx.x & 31;
  int i = blockIdx.x * 8 + (threadIdx.x >> 5);
  float vs = 0.f, vd = 0.f;
  for (int n = L; n < Nout; n += 32) {
    float x = Wh[(size_t)i * ldc + n];
    vs += x * asrc[n]; vd += x * adst[n];
  }
  #pragma unroll
  for (int off = 16; off; off >>= 1) { vs += __shfl_xor(vs, off); vd += __shfl_xor(vd, off); }
  if (L == 0) { s[i] = vs; d[i] = vd; }
}

// ---------------------------------------------------------------------------
// 4. Pass 1: per-row masked softmax stats (max m_i, 1/sum l_i), block per row
// ---------------------------------------------------------------------------
__global__ __launch_bounds__(256) void attn_stats(const float* __restrict__ s,
                                                  const float* __restrict__ d,
                                                  const unsigned* __restrict__ mask,
                                                  float* __restrict__ mo, float* __restrict__ li) {
  __shared__ float dl[NN];            // 32 KB stage of d vector
  __shared__ float wm[8], wl[8];
  int tid = threadIdx.x, i = blockIdx.x;
  for (int idx = tid; idx < NN / 4; idx += 256) ((v4f*)dl)[idx] = ((const v4f*)d)[idx];
  __syncthreads();
  float si = s[i];
  float m = -3.4e38f, l = 0.f;
  unsigned mw = mask[(size_t)i * NWORD + tid];          // this thread's 32 j's
  #pragma unroll 4
  for (int b = 0; b < 32; ++b) {
    if ((mw >> b) & 1u) {
      float e = si + dl[tid * 32 + b];
      e = e > 0.f ? e : 0.2f * e;
      if (e > m) { l *= __expf(m - e); m = e; }
      l += __expf(e - m);
    }
  }
  #pragma unroll
  for (int off = 16; off; off >>= 1) {
    float m2 = __shfl_xor(m, off), l2 = __shfl_xor(l, off);
    float mn = fmaxf(m, m2);
    l = l * __expf(m - mn) + l2 * __expf(m2 - mn);
    m = mn;
  }
  if ((tid & 31) == 0) { wm[tid >> 5] = m; wl[tid >> 5] = l; }
  __syncthreads();
  if (tid == 0) {
    float M = wm[0], Lm = wl[0];
    for (int k = 1; k < 8; ++k) {
      float mn = fmaxf(M, wm[k]);
      Lm = Lm * __expf(M - mn) + wl[k] * __expf(wm[k] - mn);
      M = mn;
    }
    mo[i] = M; li[i] = Lm > 0.f ? 1.f / Lm : 0.f;
  }
}

// ---------------------------------------------------------------------------
// 5. Pass 2: Out[i,:] = sum_j p_ij * Wh[j,:]  via WMMA.
//    Block = NT waves, 32 i-rows, j-chunks of 64: wave c owns col tile c and
//    2 accumulator subtiles; each B fragment reused for both subtiles.
// ---------------------------------------------------------------------------
template<int NT>
__global__ __launch_bounds__(NT * 32) void attn_wmma(
    const float* __restrict__ s, const float* __restrict__ d,
    const float* __restrict__ m, const float* __restrict__ li,
    const unsigned* __restrict__ mask,
    const bf16* __restrict__ Whb, float* __restrict__ Out) {
  const int ldc = NT * 16;
  __shared__ bf16  P[32][72];         // normalized exp tile, row-major [i][j], padded
  __shared__ bf16  Bs[NT * 16][72];   // Wh chunk column-major [n][k], padded
  __shared__ float sS[32], sM[32], sL[32];

  const int tid = threadIdx.x, L = tid & 31, w = tid >> 5, half = L >> 4;
  const int ibase = blockIdx.x * 32;
  if (tid < 32) { sS[tid] = s[ibase + tid]; sM[tid] = m[ibase + tid]; sL[tid] = li[ibase + tid]; }

  v8f acc[2];
  #pragma unroll
  for (int sub = 0; sub < 2; ++sub)
    #pragma unroll
    for (int e = 0; e < 8; ++e) acc[sub][e] = 0.f;
  __syncthreads();

  const bf16* arow = &P[L & 15][8 * half];               // + sub*16 rows, + kf*32 k
  const bf16* brow = &Bs[w * 16 + (L & 15)][8 * half];   // + kf*32 k

  for (int jc = 0; jc < NN / 64; ++jc) {
    const int jbase = jc << 6;
    // stage P (32 rows x 64 j): pair of adjacent j per thread -> one ds_store_b32
    for (int idx = tid; idx < 32 * 32; idx += NT * 32) {
      int jj = (idx & 31) * 2, i = idx >> 5;
      unsigned mw = mask[(size_t)(ibase + i) * NWORD + (jbase >> 5) + (jj >> 5)];
      float e0 = sS[i] + d[jbase + jj];
      float e1 = sS[i] + d[jbase + jj + 1];
      e0 = e0 > 0.f ? e0 : 0.2f * e0;
      e1 = e1 > 0.f ? e1 : 0.2f * e1;
      float p0 = ((mw >> (jj & 31)) & 1u)       ? __expf(e0 - sM[i]) * sL[i] : 0.f;
      float p1 = ((mw >> ((jj + 1) & 31)) & 1u) ? __expf(e1 - sM[i]) * sL[i] : 0.f;
      *(unsigned*)&P[i][jj] = pk2bf(p0, p1);
    }
    // stage Wh chunk [64 x ldc] -> column-major LDS, 2 bf16 per u32 store
    for (int idx = tid; idx < ldc * 32; idx += NT * 32) {
      int n = idx % ldc, k = (idx / ldc) * 2;
      union { bf16 h[2]; unsigned u; } x;
      x.h[0] = Whb[(size_t)(jbase + k) * ldc + n];
      x.h[1] = Whb[(size_t)(jbase + k + 1) * ldc + n];
      *(unsigned*)&Bs[n][k] = x.u;
    }
    __syncthreads();

    // 4 WMMAs: t = (kf<<1)|sub
    FragBF ua[2], ub[2];
    ub[0].i4[0] = *(const v4i*)(brow);
    ub[0].i4[1] = *(const v4i*)(brow + 16);
    ua[0].i4[0] = *(const v4i*)(arow);
    ua[0].i4[1] = *(const v4i*)(arow + 16);
    #pragma unroll
    for (int t = 0; t < 4; ++t) {
      int nt = t + 1;
      if (nt < 4) {
        int nkf = nt >> 1, nsub = nt & 1;
        const bf16* ap = arow + (size_t)nsub * 16 * 72 + 32 * nkf;
        ua[nt & 1].i4[0] = *(const v4i*)ap;
        ua[nt & 1].i4[1] = *(const v4i*)(ap + 16);
        if (nsub == 0) {
          const bf16* bp = brow + 32 * nkf;
          ub[nkf].i4[0] = *(const v4i*)bp;
          ub[nkf].i4[1] = *(const v4i*)(bp + 16);
        }
      }
      acc[t & 1] = __builtin_amdgcn_wmma_f32_16x16x32_bf16(
          false, ua[t & 1].v, false, ub[t >> 1].v, (short)0, acc[t & 1], false, false);
    }
    __syncthreads();
  }

  int n = w * 16 + (L & 15);
  #pragma unroll
  for (int sub = 0; sub < 2; ++sub)
    #pragma unroll
    for (int v = 0; v < 8; ++v)
      Out[(size_t)(ibase + sub * 16 + 8 * half + v) * ldc + n] = acc[sub][v];
}

// ---------------------------------------------------------------------------
// 6. log_softmax over 40 classes (wave per row), logits ld = 48 (padded)
// ---------------------------------------------------------------------------
__global__ __launch_bounds__(256) void logsoftmax40(const float* __restrict__ logits,
                                                    float* __restrict__ out) {
  int L = threadIdx.x & 31;
  int i = blockIdx.x * 8 + (threadIdx.x >> 5);
  float v0 = (L      < NCL) ? logits[(size_t)i * 48 + L]      : -3.4e38f;
  float v1 = (L + 32 < NCL) ? logits[(size_t)i * 48 + L + 32] : -3.4e38f;
  float mx = fmaxf(v0, v1);
  #pragma unroll
  for (int off = 16; off; off >>= 1) mx = fmaxf(mx, __shfl_xor(mx, off));
  float sm = ((L < NCL) ? __expf(v0 - mx) : 0.f) + ((L + 32 < NCL) ? __expf(v1 - mx) : 0.f);
  #pragma unroll
  for (int off = 16; off; off >>= 1) sm += __shfl_xor(sm, off);
  float ls = __logf(sm);
  if (L < NCL)      out[(size_t)i * NCL + L]      = v0 - mx - ls;
  if (L + 32 < NCL) out[(size_t)i * NCL + L + 32] = v1 - mx - ls;
}

// ---------------------------------------------------------------------------
extern "C" void kernel_launch(void* const* d_in, const int* in_sizes, int n_in,
                              void* d_out, int out_size, void* d_ws, size_t ws_size,
                              hipStream_t stream) {
  const float* fea  = (const float*)d_in[0];
  const int*   adj  = (const int*)d_in[1];
  const float* W_in = (const float*)d_in[2];
  const float* b_in = (const float*)d_in[3];
  const float* W1   = (const float*)d_in[4];
  const float* a1s  = (const float*)d_in[5];
  const float* a1d  = (const float*)d_in[6];
  const float* W2   = (const float*)d_in[7];
  const float* a2s  = (const float*)d_in[8];
  const float* a2d  = (const float*)d_in[9];
  const float* Wo   = (const float*)d_in[10];
  const float* aos  = (const float*)d_in[11];
  const float* aod  = (const float*)d_in[12];

  char* wp = (char*)d_ws;
  unsigned* mask = (unsigned*)wp; wp += (size_t)NN * NWORD * 4;   // 8 MB
  float* xA  = (float*)wp; wp += (size_t)NN * NH * 4;             // 4 MB
  float* xB  = (float*)wp; wp += (size_t)NN * NH * 4;             // 4 MB (also logits [NN x 48])
  float* Whf = (float*)wp; wp += (size_t)NN * NH * 4;             // 4 MB (also [NN x 48])
  bf16*  Whb = (bf16*)wp;  wp += (size_t)NN * NH * 2;             // 2 MB
  float* sv  = (float*)wp; wp += (size_t)NN * 4;
  float* dv  = (float*)wp; wp += (size_t)NN * 4;
  float* mm  = (float*)wp; wp += (size_t)NN * 4;
  float* li  = (float*)wp; wp += (size_t)NN * 4;

  // adj -> L2-resident bitmask (reads 256 MB once instead of 1.5 GB over 6 passes)
  pack_adj<<<NN * NWORD / 8, 256, 0, stream>>>(adj, mask);

  // x = fea @ W_in + b_in
  gemm_bf16<8><<<64, 256, 0, stream>>>(fea, NF, nullptr, 0, W_in, NH, b_in, xA, nullptr, NH);

  // layer 1: h = [fea, xA]
  gemm_bf16<8><<<64, 256, 0, stream>>>(fea, NF, xA, NH, W1, NH, nullptr, Whf, Whb, NH);
  rowdots<<<NN / 8, 256, 0, stream>>>(Whf, NH, NH, a1s, a1d, sv, dv);
  attn_stats<<<NN, 256, 0, stream>>>(sv, dv, mask, mm, li);
  attn_wmma<8><<<NN / 32, 256, 0, stream>>>(sv, dv, mm, li, mask, Whb, xB);

  // layer 2: h = [fea, xB]
  gemm_bf16<8><<<64, 256, 0, stream>>>(fea, NF, xB, NH, W2, NH, nullptr, Whf, Whb, NH);
  rowdots<<<NN / 8, 256, 0, stream>>>(Whf, NH, NH, a2s, a2d, sv, dv);
  attn_stats<<<NN, 256, 0, stream>>>(sv, dv, mask, mm, li);
  attn_wmma<8><<<NN / 32, 256, 0, stream>>>(sv, dv, mm, li, mask, Whb, xA);

  // output layer: h = [fea, xA], 40 classes padded to 48; logits -> xB (free)
  gemm_bf16<3><<<64, 256, 0, stream>>>(fea, NF, xA, NH, Wo, NCL, nullptr, Whf, Whb, NCL);
  rowdots<<<NN / 8, 256, 0, stream>>>(Whf, 48, NCL, aos, aod, sv, dv);
  attn_stats<<<NN, 256, 0, stream>>>(sv, dv, mask, mm, li);
  attn_wmma<3><<<NN / 32, 96, 0, stream>>>(sv, dv, mm, li, mask, Whb, xB);

  logsoftmax40<<<NN / 8, 256, 0, stream>>>(xB, (float*)d_out);
}
